// SSM_7060926234954
// MI455X (gfx1250) — matrix operand order repaired and checked by reading, not compiled, not run on hardware
//
#include <hip/hip_runtime.h>
#include <math.h>

// Problem constants (from reference)
#define B_   16
#define H_   128
#define P_   64
#define L_   8192
#define LC   128            // timesteps per chunk
#define NCH  (L_/LC)        // 64 chunks
#define USTR 136            // padded bf16 stride for transposed u (272B = 16B multiple)

typedef __attribute__((ext_vector_type(16))) __bf16 v16bf;
typedef __attribute__((ext_vector_type(8)))  __bf16 v8bf;
typedef __attribute__((ext_vector_type(2)))  __bf16 v2bf;
typedef __attribute__((ext_vector_type(8)))  float  v8f;

// Splice two aligned 16-byte LDS reads into one WMMA fragment (ds_load_b128 x2).
__device__ __forceinline__ v16bf frag16(const __bf16* p0, const __bf16* p1) {
  union { v16bf v; v8bf h[2]; } u;
  u.h[0] = *(const v8bf*)p0;
  u.h[1] = *(const v8bf*)p1;
  return u.v;
}

// ---- staging helpers ------------------------------------------------------

// Weight plane pair (f32 global, row-major) -> bf16 LDS; optionally negate
// plane 1 (stages -C_im so GEMM2 accumulates the real part directly).
__device__ __forceinline__ void stage_weights(const float* w0, const float* w1,
                                              __bf16* s0, __bf16* s1,
                                              bool neg1, int tid) {
  for (int i = tid; i < P_ * H_; i += 256) {
    s0[i] = (__bf16)w0[i];
    float v = w1[i];
    s1[i] = (__bf16)(neg1 ? -v : v);
  }
}

// u chunk (H_ x LC, row-major in HBM) -> transposed bf16 LDS sUt[l][h].
// Coalesced float4 global loads; transposed so WMMA B-fragments (K=h varying,
// N=l fixed) become contiguous 16-byte LDS loads.
__device__ __forceinline__ void stage_u_t(const float* u, __bf16* sUt,
                                          int b, int c0, int tid) {
  for (int i = tid; i < H_ * LC / 4; i += 256) {
    const int h  = i >> 5;          // 32 float4 per row (LC=128)
    const int l  = (i & 31) * 4;
    const float4 v = *(const float4*)(u + (size_t)(b * H_ + h) * L_ + c0 + l);
    sUt[(l + 0) * USTR + h] = (__bf16)v.x;
    sUt[(l + 1) * USTR + h] = (__bf16)v.y;
    sUt[(l + 2) * USTR + h] = (__bf16)v.z;
    sUt[(l + 3) * USTR + h] = (__bf16)v.w;
  }
}

// ---- GEMM1: Bu(LC x P_ layout, bf16 planes) = Bbar(P_ x H_) * u ----------
// 8 waves: wave w owns P-tile (w&3) and N-tiles (w>>2)*4 .. +3.
// Output stored transposed [l][p] so (a) the 64 scan threads are
// bank-conflict-free and (b) GEMM2 B-fragments are contiguous loads.
__device__ __forceinline__ void gemm1_wmma(const __bf16* sW0, const __bf16* sW1,
                                           const __bf16* sUt,
                                           __bf16* sXbr, __bf16* sXbi, int tid) {
  const int lane = tid & 31, w = tid >> 5;
  const int hi = lane >> 4, lp = lane & 15;
  const int mt = w & 3;
  const __bf16* aRow0 = sW0 + (mt * 16 + lp) * H_ + 8 * hi;
  const __bf16* aRow1 = sW1 + (mt * 16 + lp) * H_ + 8 * hi;
  for (int nn = 0; nn < 4; ++nn) {
    const int nt = (w >> 2) * 4 + nn;
    const __bf16* bRow = sUt + (nt * 16 + lp) * USTR + 8 * hi;
    v8f accr = {}; v8f acci = {};
    for (int kk = 0; kk < H_ / 32; ++kk) {
      const int k0 = kk * 32;
      // A fragment (16-bit 16x32): M = lane&15, K = 8*hi + (e<8 ? e : e+8)
      const v16bf a_re = frag16(aRow0 + k0, aRow0 + k0 + 16);
      const v16bf a_im = frag16(aRow1 + k0, aRow1 + k0 + 16);
      // B fragment: N = lane&15, K = 8*hi + (e<8 ? e : e+8)
      const v16bf bu   = frag16(bRow + k0, bRow + k0 + 16);
      accr = __builtin_amdgcn_wmma_f32_16x16x32_bf16(false, a_re, false, bu,
                                                     (short)0, accr, false, false);
      acci = __builtin_amdgcn_wmma_f32_16x16x32_bf16(false, a_im, false, bu,
                                                     (short)0, acci, false, false);
    }
    // C/D layout: row p = 8*hi + r, col l = lane&15. Adjacent r are adjacent p
    // in the [l][p] layout -> pack bf16 pairs, ds_store_b32.
#pragma unroll
    for (int r = 0; r < 8; r += 2) {
      const int p = mt * 16 + 8 * hi + r;
      const int l = nt * 16 + lp;
      v2bf pr = { (__bf16)accr[r], (__bf16)accr[r + 1] };
      v2bf pi = { (__bf16)acci[r], (__bf16)acci[r + 1] };
      *(v2bf*)&sXbr[l * P_ + p] = pr;
      *(v2bf*)&sXbi[l * P_ + p] = pi;
    }
  }
}

// ---- kernel 1: per-chunk carry (GEMM1 + zero-init scan, keep tail) --------

__global__ __launch_bounds__(256) void k_carry(const float* __restrict__ u,
                                               const float* __restrict__ lre,
                                               const float* __restrict__ lim,
                                               const float* __restrict__ bre,
                                               const float* __restrict__ bim,
                                               float* __restrict__ carry_re,
                                               float* __restrict__ carry_im) {
  __shared__ __bf16 sW0[P_ * H_];
  __shared__ __bf16 sW1[P_ * H_];
  __shared__ __bf16 sUt[LC * USTR];
  __shared__ __bf16 sXbr[LC * P_];
  __shared__ __bf16 sXbi[LC * P_];
  const int tid = threadIdx.x, ch = blockIdx.x, b = blockIdx.y;

  stage_weights(bre, bim, sW0, sW1, false, tid);
  stage_u_t(u, sUt, b, ch * LC, tid);
  __syncthreads();
  gemm1_wmma(sW0, sW1, sUt, sXbr, sXbi, tid);
  __syncthreads();

  if (tid < P_) {                       // diagonal recurrence, carry only
    const float lr = lre[tid], li = lim[tid];
    float xr = 0.f, xi = 0.f;
    for (int l = 0; l < LC; ++l) {
      const float br = (float)sXbr[l * P_ + tid];
      const float bi = (float)sXbi[l * P_ + tid];
      const float nr = lr * xr - li * xi + br;
      const float ni = lr * xi + li * xr + bi;
      xr = nr; xi = ni;
    }
    carry_re[(b * NCH + ch) * P_ + tid] = xr;
    carry_im[(b * NCH + ch) * P_ + tid] = xi;
  }
}

// ---- kernel 2: sequential prefix over chunk carries (tiny) ----------------

__global__ __launch_bounds__(64) void k_prefix(const float* __restrict__ lre,
                                               const float* __restrict__ lim,
                                               const float* __restrict__ carry_re,
                                               const float* __restrict__ carry_im,
                                               float* __restrict__ init_re,
                                               float* __restrict__ init_im) {
  const int p = threadIdx.x, b = blockIdx.x;
  float ar = lre[p], ai = lim[p];
  for (int i = 0; i < 7; ++i) {         // Lambda^128 via 7 complex squarings
    const float nr = ar * ar - ai * ai;
    const float ni = 2.f * ar * ai;
    ar = nr; ai = ni;
  }
  float sr = 0.f, si = 0.f;
  for (int j = 0; j < NCH; ++j) {
    init_re[(b * NCH + j) * P_ + p] = sr;
    init_im[(b * NCH + j) * P_ + p] = si;
    const float cr = carry_re[(b * NCH + j) * P_ + p];
    const float ci = carry_im[(b * NCH + j) * P_ + p];
    const float nr = ar * sr - ai * si + cr;
    const float ni = ar * si + ai * sr + ci;
    sr = nr; si = ni;
  }
}

// ---- kernel 3: GEMM1 + scan(carry-in) + GEMM2(real part) + D*u + gelu -----

__global__ __launch_bounds__(256) void k_main(const float* __restrict__ u,
                                              const float* __restrict__ lre,
                                              const float* __restrict__ lim,
                                              const float* __restrict__ bre,
                                              const float* __restrict__ bim,
                                              const float* __restrict__ cre,
                                              const float* __restrict__ cim,
                                              const float* __restrict__ Dm,
                                              const float* __restrict__ init_re,
                                              const float* __restrict__ init_im,
                                              float* __restrict__ out) {
  __shared__ __bf16 sW0[P_ * H_];
  __shared__ __bf16 sW1[P_ * H_];
  __shared__ __bf16 sUt[LC * USTR];
  __shared__ __bf16 sXbr[LC * P_];
  __shared__ __bf16 sXbi[LC * P_];
  const int tid = threadIdx.x, ch = blockIdx.x, b = blockIdx.y;
  const int c0 = ch * LC;

  stage_weights(bre, bim, sW0, sW1, false, tid);
  stage_u_t(u, sUt, b, c0, tid);
  __syncthreads();
  gemm1_wmma(sW0, sW1, sUt, sXbr, sXbi, tid);
  __syncthreads();

  if (tid < P_) {        // in-place inclusive scan; f32 state, bf16 storage
    const float lr = lre[tid], li = lim[tid];
    float xr = init_re[(b * NCH + ch) * P_ + tid];
    float xi = init_im[(b * NCH + ch) * P_ + tid];
    for (int l = 0; l < LC; ++l) {
      const float br = (float)sXbr[l * P_ + tid];
      const float bi = (float)sXbi[l * P_ + tid];
      const float nr = lr * xr - li * xi + br;
      const float ni = lr * xi + li * xr + bi;
      xr = nr; xi = ni;
      sXbr[l * P_ + tid] = (__bf16)xr;
      sXbi[l * P_ + tid] = (__bf16)xi;
    }
  }
  __syncthreads();
  stage_weights(cre, cim, sW0, sW1, true, tid);   // sW1 = -C_im
  __syncthreads();

  // GEMM2: y(H_ x LC) = C_re*x_re + (-C_im)*x_im ; wave w owns H-tile w.
  const int lane = tid & 31, w = tid >> 5;
  const int hi = lane >> 4, lp = lane & 15;
  const __bf16* aRow0 = sW0 + (w * 16 + lp) * P_ + 8 * hi;
  const __bf16* aRow1 = sW1 + (w * 16 + lp) * P_ + 8 * hi;
  float dh[8];
#pragma unroll
  for (int r = 0; r < 8; ++r) {
    const int h = w * 16 + 8 * hi + r;
    dh[r] = Dm[h * H_ + h];
  }
  for (int nt = 0; nt < LC / 16; ++nt) {
    const __bf16* xRowR = sXbr + (nt * 16 + lp) * P_ + 8 * hi;
    const __bf16* xRowI = sXbi + (nt * 16 + lp) * P_ + 8 * hi;
    v8f acc = {};
    for (int kk = 0; kk < P_ / 32; ++kk) {
      const int k0 = kk * 32;
      const v16bf a_re = frag16(aRow0 + k0, aRow0 + k0 + 16);
      const v16bf a_im = frag16(aRow1 + k0, aRow1 + k0 + 16);
      const v16bf x_re = frag16(xRowR + k0, xRowR + k0 + 16);
      const v16bf x_im = frag16(xRowI + k0, xRowI + k0 + 16);
      acc = __builtin_amdgcn_wmma_f32_16x16x32_bf16(false, a_re, false, x_re,
                                                    (short)0, acc, false, false);
      acc = __builtin_amdgcn_wmma_f32_16x16x32_bf16(false, a_im, false, x_im,
                                                    (short)0, acc, false, false);
    }
#pragma unroll
    for (int r = 0; r < 8; ++r) {
      const int h = w * 16 + 8 * hi + r;
      const int l = nt * 16 + lp;
      const float uv = (float)sUt[l * USTR + h];
      const float x  = acc[r] + dh[r] * uv;
      const float g  = 0.5f * x * (1.0f + erff(x * 0.70710678118654752f));
      out[(size_t)(b * H_ + h) * L_ + c0 + l] = g;
    }
  }
}

// ---- launch ---------------------------------------------------------------

extern "C" void kernel_launch(void* const* d_in, const int* in_sizes, int n_in,
                              void* d_out, int out_size, void* d_ws, size_t ws_size,
                              hipStream_t stream) {
  (void)in_sizes; (void)n_in; (void)out_size; (void)ws_size;
  const float* u   = (const float*)d_in[0];
  const float* lre = (const float*)d_in[1];
  const float* lim = (const float*)d_in[2];
  const float* bre = (const float*)d_in[3];
  const float* bim = (const float*)d_in[4];
  const float* cre = (const float*)d_in[5];
  const float* cim = (const float*)d_in[6];
  const float* Dm  = (const float*)d_in[7];
  float* out = (float*)d_out;

  float* ws = (float*)d_ws;
  const size_t BNP = (size_t)B_ * NCH * P_;   // 65536 floats per plane
  float* carry_re = ws;
  float* carry_im = ws + BNP;
  float* init_re  = ws + 2 * BNP;
  float* init_im  = ws + 3 * BNP;

  k_carry <<<dim3(NCH, B_), 256, 0, stream>>>(u, lre, lim, bre, bim,
                                              carry_re, carry_im);
  k_prefix<<<dim3(B_),      64, 0, stream>>>(lre, lim, carry_re, carry_im,
                                              init_re, init_im);
  k_main  <<<dim3(NCH, B_), 256, 0, stream>>>(u, lre, lim, bre, bim, cre, cim,
                                              Dm, init_re, init_im, out);
}
